// GDEncoder_81681688035908
// MI455X (gfx1250) — matrix operand order repaired and checked by reading, not compiled, not run on hardware
//
#include <hip/hip_runtime.h>
#include <hip/hip_bf16.h>

typedef __attribute__((ext_vector_type(16))) _Float16 v16h;
typedef __attribute__((ext_vector_type(8)))  _Float16 v8h;
typedef __attribute__((ext_vector_type(8)))  float    v8f;

#define TT   16
#define BB   512
#define NNN  8192
#define GGG  39
#define ENCC 128
#define HHH  4
#define DDD  32
#define RR   (BB + NNN)   // 8704 combined rows (hist first, then nbrs)

static constexpr float kScale = 0.08838834764831845f; // 1/sqrt(ENC=128)
static constexpr float kLnEps = 1e-5f;

// ---------------------------------------------------------------------------
// Wave helpers (wave32)
// ---------------------------------------------------------------------------
__device__ __forceinline__ float waveReduceSum(float v) {
  #pragma unroll
  for (int off = 16; off > 0; off >>= 1) v += __shfl_xor(v, off, 32);
  return v;
}
__device__ __forceinline__ float sigmoidf_(float x) {
  return 1.0f / (1.0f + __expf(-x));
}

// ---------------------------------------------------------------------------
// Weight fragment packing for WMMA B-operand (W is N x K row-major, C = A @ W^T)
// Fragment (nt, kt): 32 lanes x 16 halves contiguous.
//   lane l, elem e:  n = nt*16 + (l & 15),  k = kt*32 + (e & 7) + (e < 8 ? 0 : 16) + (l < 16 ? 0 : 8)
// ---------------------------------------------------------------------------
__global__ void pack_w_kernel(const float* __restrict__ W, _Float16* __restrict__ out,
                              int Nn, int K) {
  int tid = blockIdx.x * blockDim.x + threadIdx.x;
  int total = Nn * K;
  if (tid >= total) return;
  int e    = tid & 15;
  int l    = (tid >> 4) & 31;
  int frag = tid >> 9;
  int kFrags = K >> 5;
  int nt = frag / kFrags, kt = frag - nt * kFrags;
  int n = (nt << 4) + (l & 15);
  int k = (kt << 5) + (e & 7) + ((e < 8) ? 0 : 16) + ((l < 16) ? 0 : 8);
  out[tid] = (_Float16)W[(size_t)n * K + k];
}

__global__ void add_vec_kernel(const float* a, const float* b, float* o, int n) {
  int i = blockIdx.x * blockDim.x + threadIdx.x;
  if (i < n) o[i] = a[i] + b[i];
}

// ---------------------------------------------------------------------------
// Generic WMMA GEMM: C[M,Nn] = A[M,K](f16) @ Wp^T + bias + Cin
// One wave per 32x32 macro-tile (2x2 of 16x16 WMMA tiles): 2 A-frags + 2
// B-frags per K-step feed 4 x v_wmma_f32_16x16x32_f16 (register-level reuse,
// 2 b128 loads per WMMA instead of 4). All M/Nn used are multiples of 32.
// A fragment: lane l (m = l&15) holds k = kb+{0..7,16..23} (l<16) or +{8..15,24..31}.
// C layout: vgpr r, lane l -> m = r + (l<16?0:8), n = l&15 (bias is per-lane).
// ---------------------------------------------------------------------------
__device__ __forceinline__ void store_ctile(const v8f& acc, float bn, int mBase, int n,
                                            const float* __restrict__ Cin,
                                            float* __restrict__ Cout,
                                            _Float16* __restrict__ CoutH, int Nn) {
  #pragma unroll
  for (int r = 0; r < 8; ++r) {
    size_t off = (size_t)(mBase + r) * Nn + n;
    float v = acc[r] + bn + (Cin ? Cin[off] : 0.0f);
    if (Cout)  Cout[off]  = v;
    if (CoutH) CoutH[off] = (_Float16)v;
  }
}

__global__ void __launch_bounds__(256)
gemm_wmma_kernel(const _Float16* __restrict__ A, const _Float16* __restrict__ Bp,
                 const float* __restrict__ bias, const float* __restrict__ Cin,
                 float* __restrict__ Cout, _Float16* __restrict__ CoutH,
                 int M, int Nn, int K) {
  int wave = blockIdx.x * (blockDim.x >> 5) + (threadIdx.x >> 5);
  int nT2 = Nn >> 5;
  int mt = wave / nT2;
  int nt = wave - mt * nT2;
  if (mt >= (M >> 5)) return;
  int lane = threadIdx.x & 31;
  int kFrags = K >> 5;

  const _Float16* aRow0  = A + (size_t)((mt << 5) + (lane & 15)) * K + ((lane < 16) ? 0 : 8);
  const _Float16* aRow1  = aRow0 + (size_t)16 * K;
  const _Float16* bBase0 = Bp + (size_t)(2 * nt) * kFrags * 512 + lane * 16;
  const _Float16* bBase1 = bBase0 + (size_t)kFrags * 512;

  v8f acc00 = {}, acc01 = {}, acc10 = {}, acc11 = {};
  for (int kt = 0; kt < kFrags; ++kt) {
    v8h a0lo = *(const v8h*)(aRow0 + kt * 32);
    v8h a0hi = *(const v8h*)(aRow0 + kt * 32 + 16);
    v8h a1lo = *(const v8h*)(aRow1 + kt * 32);
    v8h a1hi = *(const v8h*)(aRow1 + kt * 32 + 16);
    v16h a0 = __builtin_shufflevector(a0lo, a0hi, 0,1,2,3,4,5,6,7,8,9,10,11,12,13,14,15);
    v16h a1 = __builtin_shufflevector(a1lo, a1hi, 0,1,2,3,4,5,6,7,8,9,10,11,12,13,14,15);
    v16h b0 = *(const v16h*)(bBase0 + (size_t)kt * 512);
    v16h b1 = *(const v16h*)(bBase1 + (size_t)kt * 512);
    acc00 = __builtin_amdgcn_wmma_f32_16x16x32_f16(false, a0, false, b0, (short)0, acc00, false, false);
    acc01 = __builtin_amdgcn_wmma_f32_16x16x32_f16(false, a0, false, b1, (short)0, acc01, false, false);
    acc10 = __builtin_amdgcn_wmma_f32_16x16x32_f16(false, a1, false, b0, (short)0, acc10, false, false);
    acc11 = __builtin_amdgcn_wmma_f32_16x16x32_f16(false, a1, false, b1, (short)0, acc11, false, false);
  }

  int n0 = (nt << 5) + (lane & 15);
  int n1 = n0 + 16;
  float bn0 = bias ? bias[n0] : 0.0f;
  float bn1 = bias ? bias[n1] : 0.0f;
  int m0 = (mt << 5) + ((lane < 16) ? 0 : 8);
  int m1 = m0 + 16;
  store_ctile(acc00, bn0, m0, n0, Cin, Cout, CoutH, Nn);
  store_ctile(acc01, bn1, m0, n1, Cin, Cout, CoutH, Nn);
  store_ctile(acc10, bn0, m1, n0, Cin, Cout, CoutH, Nn);
  store_ctile(acc11, bn1, m1, n1, Cin, Cout, CoutH, Nn);
}

// ---------------------------------------------------------------------------
// Feature concat + elu(x @ W1^T + b1) for combined rows (hist rows 0..B-1, nbr rows B..R-1)
// ---------------------------------------------------------------------------
__global__ void encode_kernel(const float* __restrict__ hist, const float* __restrict__ cls,
                              const float* __restrict__ va,   const float* __restrict__ nbrs,
                              const float* __restrict__ ncls, const float* __restrict__ nva,
                              const float* __restrict__ W1,   const float* __restrict__ b1,
                              _Float16* __restrict__ enc) {
  size_t idx = (size_t)blockIdx.x * blockDim.x + threadIdx.x;
  size_t total = (size_t)TT * RR * 64;
  if (idx >= total) return;
  int u = idx & 63;
  size_t tr = idx >> 6;
  int r = (int)(tr % RR);
  int t = (int)(tr / RR);
  float x0, x1, x2, x3, x4;
  if (r < BB) {
    size_t o2 = ((size_t)t * BB + r) * 2, o1 = (size_t)t * BB + r;
    x0 = hist[o2]; x1 = hist[o2 + 1]; x2 = cls[o1]; x3 = va[o2]; x4 = va[o2 + 1];
  } else {
    int j = r - BB;
    size_t o2 = ((size_t)t * NNN + j) * 2, o1 = (size_t)t * NNN + j;
    x0 = nbrs[o2]; x1 = nbrs[o2 + 1]; x2 = ncls[o1]; x3 = nva[o2]; x4 = nva[o2 + 1];
  }
  const float* w = W1 + u * 5;
  float s = w[0]*x0 + w[1]*x1 + w[2]*x2 + w[3]*x3 + w[4]*x4 + b1[u];
  float e = (s > 0.0f) ? s : (__expf(s) - 1.0f);
  enc[idx] = (_Float16)e;
}

// ---------------------------------------------------------------------------
// LSTM cell pointwise update. gates = [i|f|g|o] each 128 wide.
// ---------------------------------------------------------------------------
__global__ void lstm_cell_kernel(const float* __restrict__ gates, float* __restrict__ c,
                                 _Float16* __restrict__ h_out, float* __restrict__ hist_out) {
  int idx = blockIdx.x * blockDim.x + threadIdx.x;
  if (idx >= RR * ENCC) return;
  int r = idx >> 7, u = idx & 127;
  const float* gr = gates + (size_t)r * 512;
  float ig = gr[u], fg = gr[128 + u], gg = gr[256 + u], og = gr[384 + u];
  float cc = sigmoidf_(fg) * c[idx] + sigmoidf_(ig) * tanhf(gg);
  c[idx] = cc;
  float hh = sigmoidf_(og) * tanhf(cc);
  h_out[idx] = (_Float16)hh;
  if (r < BB) hist_out[(size_t)r * ENCC + u] = hh;
}

// ---------------------------------------------------------------------------
// Prefix scan over mask cells -> neighbor index per (b,g); -1 for empty cells.
// Single block, chunked wave scan.
// ---------------------------------------------------------------------------
__global__ void scan_kernel(const unsigned char* __restrict__ mask, int* __restrict__ nbr_idx,
                            int P) {
  __shared__ int waveSums[8];
  __shared__ int chunkBase;
  if (threadIdx.x == 0) chunkBase = 0;
  __syncthreads();
  for (int base = 0; base < P; base += blockDim.x) {
    int p = base + threadIdx.x;
    int flag = (p < P) ? (mask[(size_t)p * ENCC] ? 1 : 0) : 0;
    int v = flag;
    #pragma unroll
    for (int off = 1; off < 32; off <<= 1) {
      int u = __shfl_up(v, off, 32);
      if ((threadIdx.x & 31) >= off) v += u;
    }
    int wid = threadIdx.x >> 5;
    if ((threadIdx.x & 31) == 31) waveSums[wid] = v;
    __syncthreads();
    int waveOff = 0;
    for (int w = 0; w < wid; ++w) waveOff += waveSums[w];
    int incl = chunkBase + waveOff + v;
    if (p < P) nbr_idx[p] = flag ? (incl - 1) : -1;
    __syncthreads();
    if (threadIdx.x == blockDim.x - 1) chunkBase = incl;
    __syncthreads();
  }
}

// ---------------------------------------------------------------------------
// Spatial attention over G=39 grid cells, one wave per (t,b,h); lane = d.
// Empty cells contribute k=bk, v=bv (matches reference bias-on-zero soc).
// Online (flash-style) softmax.
// ---------------------------------------------------------------------------
__global__ void __launch_bounds__(256)
spatial_attn_kernel(const float* __restrict__ q, const _Float16* __restrict__ kv,
                    const int* __restrict__ nbr_idx, const float* __restrict__ bk,
                    const float* __restrict__ bv, _Float16* __restrict__ spa) {
  int wave = blockIdx.x * 8 + (threadIdx.x >> 5);
  if (wave >= TT * BB * HHH) return;
  int h = wave % HHH;
  int tb = wave / HHH;
  int b = tb % BB, t = tb / BB;
  int lane = threadIdx.x & 31;
  float qd = q[((size_t)t * BB + b) * ENCC + h * 32 + lane];
  const int* cellIdx = nbr_idx + b * GGG;
  float mrun = -1e30f, ssum = 0.0f, acc = 0.0f;
  for (int g = 0; g < GGG; ++g) {
    int j = cellIdx[g];
    float kd, vd;
    if (j >= 0) {
      const _Float16* kvr = kv + ((size_t)t * NNN + j) * 256 + h * 32 + lane;
      kd = (float)kvr[0];
      vd = (float)kvr[128];
    } else {
      kd = bk[h * 32 + lane];
      vd = bv[h * 32 + lane];
    }
    float s = waveReduceSum(qd * kd) * kScale;
    float mnew = fmaxf(mrun, s);
    float sc = __expf(mrun - mnew);
    float e  = __expf(s - mnew);
    acc  = acc * sc + e * vd;
    ssum = ssum * sc + e;
    mrun = mnew;
  }
  spa[((size_t)t * BB + b) * ENCC + h * 32 + lane] = (_Float16)(acc / ssum);
}

// ---------------------------------------------------------------------------
// Temporal attention over T=16, one wave per (t,b,h); lane = d.
// qkvt rows (t*B+b) hold [qt(128)|kt(128)|vt(128)].
// ---------------------------------------------------------------------------
__global__ void __launch_bounds__(256)
temporal_attn_kernel(const float* __restrict__ qkvt, _Float16* __restrict__ tmp) {
  int wave = blockIdx.x * 8 + (threadIdx.x >> 5);
  if (wave >= TT * BB * HHH) return;
  int h = wave % HHH;
  int tb = wave / HHH;
  int b = tb % BB, t = tb / BB;
  int lane = threadIdx.x & 31;
  float qd = qkvt[((size_t)t * BB + b) * 384 + h * 32 + lane];
  float mrun = -1e30f, ssum = 0.0f, acc = 0.0f;
  for (int s = 0; s < TT; ++s) {
    const float* row = qkvt + ((size_t)s * BB + b) * 384;
    float kd = row[128 + h * 32 + lane];
    float vd = row[256 + h * 32 + lane];
    float sc0 = waveReduceSum(qd * kd) * kScale;
    float mnew = fmaxf(mrun, sc0);
    float sc = __expf(mrun - mnew);
    float e  = __expf(sc0 - mnew);
    acc  = acc * sc + e * vd;
    ssum = ssum * sc + e;
    mrun = mnew;
  }
  tmp[((size_t)t * BB + b) * ENCC + h * 32 + lane] = (_Float16)(acc / ssum);
}

// ---------------------------------------------------------------------------
// GLU + residual + LayerNorm. One wave per row; 4 elems/lane.
// s1 = resid + glu_a * sigmoid(glu_g);  values = LN(s1) (f16 for next GEMM).
// ---------------------------------------------------------------------------
__global__ void __launch_bounds__(256)
glu_ln_kernel(const float* __restrict__ glu, const float* __restrict__ resid,
              const float* __restrict__ ln_g, const float* __restrict__ ln_b,
              float* __restrict__ s1, _Float16* __restrict__ values, int rows) {
  int row = blockIdx.x * 8 + (threadIdx.x >> 5);
  if (row >= rows) return;
  int lane = threadIdx.x & 31;
  float x[4], mean = 0.0f;
  #pragma unroll
  for (int i = 0; i < 4; ++i) {
    int d = lane + 32 * i;
    float a = glu[(size_t)row * 256 + d];
    float g = glu[(size_t)row * 256 + 128 + d];
    float v = resid[(size_t)row * ENCC + d] + a * sigmoidf_(g);
    x[i] = v; mean += v;
  }
  mean = waveReduceSum(mean) * (1.0f / ENCC);
  float var = 0.0f;
  #pragma unroll
  for (int i = 0; i < 4; ++i) { float dl = x[i] - mean; var += dl * dl; }
  var = waveReduceSum(var) * (1.0f / ENCC);
  float rstd = rsqrtf(var + kLnEps);
  #pragma unroll
  for (int i = 0; i < 4; ++i) {
    int d = lane + 32 * i;
    float y = (x[i] - mean) * rstd * ln_g[d] + ln_b[d];
    s1[(size_t)row * ENCC + d] = x[i];
    values[(size_t)row * ENCC + d] = (_Float16)y;
  }
}

// ---------------------------------------------------------------------------
// Final: out = LN(s1 + glu2_a * sigmoid(glu2_g)), written transposed (b,t,:).
// ---------------------------------------------------------------------------
__global__ void __launch_bounds__(256)
final_kernel(const float* __restrict__ glu2, const float* __restrict__ s1,
             const float* __restrict__ ln_g, const float* __restrict__ ln_b,
             float* __restrict__ out) {
  int row = blockIdx.x * 8 + (threadIdx.x >> 5);
  if (row >= TT * BB) return;
  int b = row % BB, t = row / BB;
  int lane = threadIdx.x & 31;
  float x[4], mean = 0.0f;
  #pragma unroll
  for (int i = 0; i < 4; ++i) {
    int d = lane + 32 * i;
    float a = glu2[(size_t)row * 256 + d];
    float g = glu2[(size_t)row * 256 + 128 + d];
    float v = s1[(size_t)row * ENCC + d] + a * sigmoidf_(g);
    x[i] = v; mean += v;
  }
  mean = waveReduceSum(mean) * (1.0f / ENCC);
  float var = 0.0f;
  #pragma unroll
  for (int i = 0; i < 4; ++i) { float dl = x[i] - mean; var += dl * dl; }
  var = waveReduceSum(var) * (1.0f / ENCC);
  float rstd = rsqrtf(var + kLnEps);
  #pragma unroll
  for (int i = 0; i < 4; ++i) {
    int d = lane + 32 * i;
    out[((size_t)b * TT + t) * ENCC + d] = (x[i] - mean) * rstd * ln_g[d] + ln_b[d];
  }
}

// ---------------------------------------------------------------------------
// Host orchestration
// ---------------------------------------------------------------------------
extern "C" void kernel_launch(void* const* d_in, const int* in_sizes, int n_in,
                              void* d_out, int out_size, void* d_ws, size_t ws_size,
                              hipStream_t stream) {
  (void)in_sizes; (void)n_in; (void)out_size; (void)ws_size;

  const float* hist    = (const float*)d_in[0];
  const float* nbrs    = (const float*)d_in[1];
  const float* va      = (const float*)d_in[2];
  const float* nbrsva  = (const float*)d_in[3];
  const float* cls     = (const float*)d_in[6];
  const float* nbrscls = (const float*)d_in[7];
  const unsigned char* mask = (const unsigned char*)d_in[8];
  const float* W1  = (const float*)d_in[9];
  const float* b1  = (const float*)d_in[10];
  const float* Wih = (const float*)d_in[11];
  const float* Whh = (const float*)d_in[12];
  const float* bih = (const float*)d_in[13];
  const float* bhh = (const float*)d_in[14];
  const float* Wq  = (const float*)d_in[15];
  const float* bq  = (const float*)d_in[16];
  const float* Wk  = (const float*)d_in[17];
  const float* bk  = (const float*)d_in[18];
  const float* Wv  = (const float*)d_in[19];
  const float* bv  = (const float*)d_in[20];
  const float* Wg1a = (const float*)d_in[21];
  const float* bg1a = (const float*)d_in[22];
  const float* Wg1g = (const float*)d_in[23];
  const float* bg1g = (const float*)d_in[24];
  const float* Wqt = (const float*)d_in[25];
  const float* bqt = (const float*)d_in[26];
  const float* Wkt = (const float*)d_in[27];
  const float* bkt = (const float*)d_in[28];
  const float* Wvt = (const float*)d_in[29];
  const float* bvt = (const float*)d_in[30];
  const float* Wg2a = (const float*)d_in[31];
  const float* bg2a = (const float*)d_in[32];
  const float* Wg2g = (const float*)d_in[33];
  const float* bg2g = (const float*)d_in[34];
  const float* ln_g = (const float*)d_in[35];
  const float* ln_b = (const float*)d_in[36];

  char* ws = (char*)d_ws;
  size_t cur = 0;
  auto alloc = [&](size_t bytes) -> char* {
    char* p = ws + cur;
    cur += (bytes + 255) & ~(size_t)255;
    return p;
  };

  _Float16* enc_f16   = (_Float16*)alloc((size_t)TT * RR * 64 * 2);
  float*    gates     = (float*)   alloc((size_t)RR * 512 * 4);
  float*    cstate    = (float*)   alloc((size_t)RR * ENCC * 4);
  _Float16* h0        = (_Float16*)alloc((size_t)RR * ENCC * 2);
  _Float16* hs_f16    = (_Float16*)alloc((size_t)TT * RR * ENCC * 2);
  float*    hist_h    = (float*)   alloc((size_t)TT * BB * ENCC * 4);
  float*    q_f32     = (float*)   alloc((size_t)TT * BB * ENCC * 4);
  _Float16* kv_f16    = (_Float16*)alloc((size_t)TT * NNN * 256 * 2);
  int*      nbr_idx   = (int*)     alloc((size_t)BB * GGG * 4);
  _Float16* spa_f16   = (_Float16*)alloc((size_t)TT * BB * ENCC * 2);
  float*    glu1      = (float*)   alloc((size_t)TT * BB * 256 * 4);
  float*    s1        = (float*)   alloc((size_t)TT * BB * ENCC * 4);
  _Float16* values16  = (_Float16*)alloc((size_t)TT * BB * ENCC * 2);
  float*    qkvt      = (float*)   alloc((size_t)TT * BB * 384 * 4);
  _Float16* tmp_f16   = (_Float16*)alloc((size_t)TT * BB * ENCC * 2);
  float*    glu2      = (float*)   alloc((size_t)TT * BB * 256 * 4);

  _Float16* Wih_p  = (_Float16*)alloc(512 * 64 * 2);
  _Float16* Whh_p  = (_Float16*)alloc(512 * 128 * 2);
  _Float16* Wq_p   = (_Float16*)alloc(128 * 128 * 2);
  float*    Wkv_f  = (float*)   alloc(256 * 128 * 4);
  _Float16* Wkv_p  = (_Float16*)alloc(256 * 128 * 2);
  float*    Wqkvt_f= (float*)   alloc(384 * 128 * 4);
  _Float16* Wqkvt_p= (_Float16*)alloc(384 * 128 * 2);
  float*    Wg1_f  = (float*)   alloc(256 * 128 * 4);
  _Float16* Wg1_p  = (_Float16*)alloc(256 * 128 * 2);
  float*    Wg2_f  = (float*)   alloc(256 * 128 * 4);
  _Float16* Wg2_p  = (_Float16*)alloc(256 * 128 * 2);
  float*    bihhh  = (float*)alloc(512 * 4);
  float*    bkv    = (float*)alloc(256 * 4);
  float*    bqkvt  = (float*)alloc(384 * 4);
  float*    bg1    = (float*)alloc(256 * 4);
  float*    bg2    = (float*)alloc(256 * 4);

  const size_t WSZ = (size_t)128 * 128 * 4; // one 128x128 weight in bytes
  // Weight/bias concats (capture-safe D2D copies).
  hipMemcpyAsync(Wkv_f,            Wk,  WSZ, hipMemcpyDeviceToDevice, stream);
  hipMemcpyAsync(Wkv_f + 128*128,  Wv,  WSZ, hipMemcpyDeviceToDevice, stream);
  hipMemcpyAsync(Wqkvt_f,            Wqt, WSZ, hipMemcpyDeviceToDevice, stream);
  hipMemcpyAsync(Wqkvt_f + 128*128,  Wkt, WSZ, hipMemcpyDeviceToDevice, stream);
  hipMemcpyAsync(Wqkvt_f + 2*128*128,Wvt, WSZ, hipMemcpyDeviceToDevice, stream);
  hipMemcpyAsync(Wg1_f,            Wg1a, WSZ, hipMemcpyDeviceToDevice, stream);
  hipMemcpyAsync(Wg1_f + 128*128,  Wg1g, WSZ, hipMemcpyDeviceToDevice, stream);
  hipMemcpyAsync(Wg2_f,            Wg2a, WSZ, hipMemcpyDeviceToDevice, stream);
  hipMemcpyAsync(Wg2_f + 128*128,  Wg2g, WSZ, hipMemcpyDeviceToDevice, stream);
  hipMemcpyAsync(bkv,        bk,  128 * 4, hipMemcpyDeviceToDevice, stream);
  hipMemcpyAsync(bkv + 128,  bv,  128 * 4, hipMemcpyDeviceToDevice, stream);
  hipMemcpyAsync(bqkvt,        bqt, 128 * 4, hipMemcpyDeviceToDevice, stream);
  hipMemcpyAsync(bqkvt + 128,  bkt, 128 * 4, hipMemcpyDeviceToDevice, stream);
  hipMemcpyAsync(bqkvt + 256,  bvt, 128 * 4, hipMemcpyDeviceToDevice, stream);
  hipMemcpyAsync(bg1,        bg1a, 128 * 4, hipMemcpyDeviceToDevice, stream);
  hipMemcpyAsync(bg1 + 128,  bg1g, 128 * 4, hipMemcpyDeviceToDevice, stream);
  hipMemcpyAsync(bg2,        bg2a, 128 * 4, hipMemcpyDeviceToDevice, stream);
  hipMemcpyAsync(bg2 + 128,  bg2g, 128 * 4, hipMemcpyDeviceToDevice, stream);

  add_vec_kernel<<<2, 256, 0, stream>>>(bih, bhh, bihhh, 512);

  auto pack = [&](const float* src, _Float16* dst, int Nn, int K) {
    int total = Nn * K;
    pack_w_kernel<<<(total + 255) / 256, 256, 0, stream>>>(src, dst, Nn, K);
  };
  pack(Wih,     Wih_p,   512, 64);
  pack(Whh,     Whh_p,   512, 128);
  pack(Wq,      Wq_p,    128, 128);
  pack(Wkv_f,   Wkv_p,   256, 128);
  pack(Wqkvt_f, Wqkvt_p, 384, 128);
  pack(Wg1_f,   Wg1_p,   256, 128);
  pack(Wg2_f,   Wg2_p,   256, 128);

  auto gemm = [&](const _Float16* A, const _Float16* Bp, const float* bias,
                  const float* Cin, float* Cout, _Float16* CoutH,
                  int M, int Nn, int K) {
    int waves = (M >> 5) * (Nn >> 5);   // 32x32 macro-tile per wave
    gemm_wmma_kernel<<<(waves + 7) / 8, 256, 0, stream>>>(A, Bp, bias, Cin, Cout,
                                                          CoutH, M, Nn, K);
  };

  // Encoder (feature concat + elu)
  {
    size_t total = (size_t)TT * RR * 64;
    encode_kernel<<<(unsigned)((total + 255) / 256), 256, 0, stream>>>(
        hist, cls, va, nbrs, nbrscls, nbrsva, W1, b1, enc_f16);
  }

  // LSTM state init
  hipMemsetAsync(cstate, 0, (size_t)RR * ENCC * 4, stream);
  hipMemsetAsync(h0,     0, (size_t)RR * ENCC * 2, stream);

  // Combined (hist+nbrs) LSTM over T steps: shared Wih/Whh weights.
  for (int t = 0; t < TT; ++t) {
    gemm(enc_f16 + (size_t)t * RR * 64, Wih_p, bihhh, nullptr, gates, nullptr,
         RR, 512, 64);
    const _Float16* hprev = (t == 0) ? h0 : (hs_f16 + (size_t)(t - 1) * RR * ENCC);
    gemm(hprev, Whh_p, nullptr, gates, gates, nullptr, RR, 512, 128);
    lstm_cell_kernel<<<(RR * ENCC + 255) / 256, 256, 0, stream>>>(
        gates, cstate, hs_f16 + (size_t)t * RR * ENCC, hist_h + (size_t)t * BB * ENCC);
  }

  // q on hist rows, k|v on nbr rows (per-t slices of combined hidden state).
  for (int t = 0; t < TT; ++t) {
    gemm(hs_f16 + (size_t)t * RR * ENCC, Wq_p, bq, nullptr,
         q_f32 + (size_t)t * BB * ENCC, nullptr, BB, 128, 128);
    gemm(hs_f16 + ((size_t)t * RR + BB) * ENCC, Wkv_p, bkv, nullptr,
         nullptr, kv_f16 + (size_t)t * NNN * 256, NNN, 256, 128);
  }

  // Grid-cell -> neighbor index scan (bijective gather).
  scan_kernel<<<1, 256, 0, stream>>>(mask, nbr_idx, BB * GGG);

  // Spatial attention over G cells.
  spatial_attn_kernel<<<(TT * BB * HHH) / 8, 256, 0, stream>>>(
      q_f32, kv_f16, nbr_idx, bk, bv, spa_f16);

  // GLU1 + residual + LN -> s1, values
  gemm(spa_f16, Wg1_p, bg1, nullptr, glu1, nullptr, TT * BB, 256, 128);
  glu_ln_kernel<<<(TT * BB) / 8, 256, 0, stream>>>(glu1, hist_h, ln_g, ln_b,
                                                   s1, values16, TT * BB);

  // Temporal attention path
  gemm(values16, Wqkvt_p, bqkvt, nullptr, qkvt, nullptr, TT * BB, 384, 128);
  temporal_attn_kernel<<<(TT * BB * HHH) / 8, 256, 0, stream>>>(qkvt, tmp_f16);

  // GLU2 + residual + final LN -> out (B,T,ENC)
  gemm(tmp_f16, Wg2_p, bg2, nullptr, glu2, nullptr, TT * BB, 256, 128);
  final_kernel<<<(TT * BB) / 8, 256, 0, stream>>>(glu2, s1, ln_g, ln_b,
                                                  (float*)d_out);
}
